// GraphUpdate_63290638074189
// MI455X (gfx1250) — matrix operand order repaired and checked by reading, not compiled, not run on hardware
//
#include <hip/hip_runtime.h>
#include <hip/hip_bf16.h>
#include <math.h>

#define BS       2
#define NUMV     2048
#define CDIM     256
#define NHEAD    8
#define DHEAD    32
#define NUM_LAYER 4

typedef __bf16 bf16_t;
typedef bf16_t v16bf __attribute__((ext_vector_type(16)));
typedef bf16_t v8bf  __attribute__((ext_vector_type(8)));
typedef float  v8f   __attribute__((ext_vector_type(8)));

typedef unsigned int u32x4 __attribute__((ext_vector_type(4)));
typedef int          i32x4 __attribute__((ext_vector_type(4)));
typedef int          i32x8 __attribute__((ext_vector_type(8)));

union V16 { v16bf v; v8bf h[2]; };

// ---------------------------------------------------------------------------
// Tensor Data Mover: DMA a 2-D tile (tile0 x tile1 elements, row stride
// stride0 elements) from global memory into LDS. D# layout per CDNA5 ISA
// (CDNA5 ISA). Issue from ONE wave; pairs with
// s_wait_tensorcnt + workgroup barrier before other waves read the LDS.
// ---------------------------------------------------------------------------
__device__ __forceinline__ void tdm_load_2d(unsigned lds_off, const void* gptr,
                                            unsigned dsize_code,  // 0=1B,1=2B,2=4B
                                            unsigned tdim0, unsigned tdim1,
                                            unsigned tile0, unsigned tile1,
                                            unsigned stride0) {
  unsigned long long ga = (unsigned long long)(uintptr_t)gptr;
  u32x4 g0;
  g0[0] = 1u;                                          // count=1 (valid descriptor)
  g0[1] = lds_off;                                     // lds_addr [63:32]
  g0[2] = (unsigned)(ga & 0xFFFFFFFFu);                // global_addr lo
  g0[3] = (unsigned)((ga >> 32) & 0x01FFFFFFu)         // global_addr hi (57-bit)
        | (2u << 30);                                  // type=2 ("image")
  i32x8 g1;
  g1[0] = (int)(dsize_code << 16);                     // data_size @ [17:16], mask=0
  g1[1] = (int)((tdim0 & 0xFFFFu) << 16);              // tensor_dim0[15:0] @ bits 63:48
  g1[2] = (int)((tdim0 >> 16) | ((tdim1 & 0xFFFFu) << 16)); // dim0 hi / dim1 lo
  g1[3] = (int)((tdim1 >> 16) | ((tile0 & 0xFFFFu) << 16)); // dim1 hi / tile_dim0
  g1[4] = (int)(tile1 & 0xFFFFu);                      // tile_dim1 (tile_dim2=0)
  g1[5] = (int)stride0;                                // tensor_dim0_stride lo32
  g1[6] = 0;
  g1[7] = 0;
  i32x4 z4 = {0, 0, 0, 0};
#if defined(__clang_major__) && __clang_major__ >= 23
  i32x8 z8 = {0, 0, 0, 0, 0, 0, 0, 0};
  __builtin_amdgcn_tensor_load_to_lds(g0, g1, z4, z4, z8, 0);
#else
  __builtin_amdgcn_tensor_load_to_lds(g0, g1, z4, z4, 0);
#endif
}

// ---------------------------------------------------------------------------
// 1) Transpose + convert feats: (bs, C, num) f32  ->  (bs, num, C) bf16
// ---------------------------------------------------------------------------
__global__ __launch_bounds__(256) void k_transpose_bf16(
    const float* __restrict__ vin, const float* __restrict__ ein,
    bf16_t* __restrict__ vTo, bf16_t* __restrict__ eTo) {
  __shared__ float tile[32][33];
  int z = blockIdx.z;                // 0..3
  int b = z & 1, sel = z >> 1;
  const float* src = (sel ? ein : vin) + (size_t)b * CDIM * NUMV;
  bf16_t*      dst = (sel ? eTo : vTo) + (size_t)b * NUMV * CDIM;
  int n0 = blockIdx.x * 32;          // along num
  int c0 = blockIdx.y * 32;          // along C
  int tx = threadIdx.x, ty = threadIdx.y;
  #pragma unroll
  for (int j = 0; j < 32; j += 8)
    tile[ty + j][tx] = src[(size_t)(c0 + ty + j) * NUMV + n0 + tx];
  __syncthreads();
  #pragma unroll
  for (int j = 0; j < 32; j += 8)
    dst[(size_t)(n0 + ty + j) * CDIM + c0 + tx] = (bf16_t)tile[tx][ty + j];
}

// ---------------------------------------------------------------------------
// 2) Convert weights to bf16 (row-major kept: row = c_out, contiguous along c)
// ---------------------------------------------------------------------------
__global__ __launch_bounds__(256) void k_convw(
    const float* __restrict__ Wq, const float* __restrict__ Wk,
    bf16_t* __restrict__ Wqb, bf16_t* __restrict__ Wkb) {
  int idx = blockIdx.x * 256 + threadIdx.x;
  if (idx < CDIM * CDIM) Wqb[idx] = (bf16_t)Wq[idx];
  else                   Wkb[idx - CDIM * CDIM] = (bf16_t)Wk[idx - CDIM * CDIM];
}

// ---------------------------------------------------------------------------
// 3) Projection GEMM via WMMA: outT[n, cout] = sum_c featT[n,c]*W[cout,c]+bias
// ---------------------------------------------------------------------------
__global__ __launch_bounds__(256) void k_proj(
    const bf16_t* __restrict__ vT, const bf16_t* __restrict__ eT,
    const bf16_t* __restrict__ Wqb, const bf16_t* __restrict__ Wkb,
    const float* __restrict__ bq, const float* __restrict__ bk,
    bf16_t* __restrict__ QT, bf16_t* __restrict__ KT) {
  int z = blockIdx.z;                // 0..3
  int b = z & 1, sel = z >> 1;
  const bf16_t* featT = (sel ? eT : vT) + (size_t)b * NUMV * CDIM;
  const bf16_t* Wb    = sel ? Wkb : Wqb;
  const float*  bias  = sel ? bk : bq;
  bf16_t*       outT  = (sel ? KT : QT) + (size_t)b * NUMV * CDIM;

  int wave = threadIdx.x >> 5, lane = threadIdx.x & 31;
  int m = lane & 15, half = lane >> 4;
  int n0    = blockIdx.x * 16;                 // rows (num dim) = M
  int cout0 = (blockIdx.y * 8 + wave) * 16;    // out-channel tile = N
  const bf16_t* arow = featT + (size_t)(n0 + m) * CDIM;      // A: lane m = row M
  const bf16_t* brow = Wb    + (size_t)(cout0 + m) * CDIM;   // B: lane m = col N

  v8f acc = {0.f,0.f,0.f,0.f,0.f,0.f,0.f,0.f};
  #pragma unroll
  for (int kc = 0; kc < CDIM; kc += 32) {
    V16 A, B;
    A.h[0] = *(const v8bf*)(arow + kc + 8 * half);
    A.h[1] = *(const v8bf*)(arow + kc + 16 + 8 * half);
    B.h[0] = *(const v8bf*)(brow + kc + 8 * half);
    B.h[1] = *(const v8bf*)(brow + kc + 16 + 8 * half);
    acc = __builtin_amdgcn_wmma_f32_16x16x32_bf16(
        false, A.v, false, B.v, (short)0, acc, false, false);
  }
  float bv = bias[cout0 + m];
  #pragma unroll
  for (int r = 0; r < 8; ++r) {
    int nrow = n0 + r + 8 * half;   // C/D layout: lanes16-31 hold M = r+8
    outT[(size_t)nrow * CDIM + cout0 + m] = (bf16_t)(acc[r] + bv);
  }
}

// ---------------------------------------------------------------------------
// 4) Attention + masked sigmoid head-average -> score (bs, num, num) f32
//    A-tile (Q rows, shared by all 8 waves of the block) is staged into LDS
//    once via the Tensor Data Mover; WMMA A-operand then comes from LDS.
// ---------------------------------------------------------------------------
__global__ __launch_bounds__(256) void k_attn(
    const bf16_t* __restrict__ QT, const bf16_t* __restrict__ KT,
    const float* __restrict__ H0, float* __restrict__ score) {
  __shared__ __align__(16) bf16_t qtile[16 * CDIM];   // 8 KB

  int b = blockIdx.z;
  int wave = threadIdx.x >> 5, lane = threadIdx.x & 31;
  int m = lane & 15, half = lane >> 4;
  int o0 = blockIdx.x * 16;
  int i0 = (blockIdx.y * 8 + wave) * 16;

  // One wave DMAs the 16x256 bf16 Q panel for this block into LDS.
  if (wave == 0) {
    const bf16_t* qsrc = QT + ((size_t)b * NUMV + o0) * CDIM;
    tdm_load_2d((unsigned)(uintptr_t)(&qtile[0]), qsrc,
                /*dsize=2B*/1, /*tdim0*/CDIM, /*tdim1*/NUMV,
                /*tile0*/CDIM, /*tile1*/16, /*stride0*/CDIM);
    __builtin_amdgcn_s_wait_tensorcnt(0);
  }
  __syncthreads();

  const bf16_t* qrow = qtile + (size_t)m * CDIM;                       // LDS
  const bf16_t* krow = KT + ((size_t)b * NUMV + i0 + m) * CDIM;        // global

  v8f acc[NHEAD];
  #pragma unroll
  for (int h = 0; h < NHEAD; ++h) acc[h] = (v8f){0.f,0.f,0.f,0.f,0.f,0.f,0.f,0.f};

  #pragma unroll
  for (int h = 0; h < NHEAD; ++h) {
    V16 A, B;
    const bf16_t* pa = qrow + h * DHEAD;
    const bf16_t* pb = krow + h * DHEAD;
    A.h[0] = *(const v8bf*)(pa + 8 * half);        // ds_load_b128
    A.h[1] = *(const v8bf*)(pa + 16 + 8 * half);
    B.h[0] = *(const v8bf*)(pb + 8 * half);        // global_load_b128
    B.h[1] = *(const v8bf*)(pb + 16 + 8 * half);
    acc[h] = __builtin_amdgcn_wmma_f32_16x16x32_bf16(
        false, A.v, false, B.v, (short)0, acc[h], false, false);
  }

  const float scale = 0.17677669529663687f;  // 1/sqrt(32)
  int icol = i0 + m;
  #pragma unroll
  for (int r = 0; r < 8; ++r) {
    int orow = o0 + r + 8 * half;
    size_t idx = ((size_t)b * NUMV + orow) * NUMV + icol;
    float h0 = H0[idx];
    float madd = ((1.0f - h0) != 0.0f) ? -1.0e9f : 0.0f;
    float s = 0.0f;
    #pragma unroll
    for (int h = 0; h < NHEAD; ++h) {
      float x = acc[h][r] * scale + madd;
      s += 1.0f / (1.0f + __expf(-x));
    }
    score[idx] = s * (1.0f / NHEAD);
  }
}

// ---------------------------------------------------------------------------
// 5) Exact kth-largest per row via 31-step bisection on float bit patterns
//    (scores >= 0, so uint order == float order). Row staged in LDS via TDM.
// ---------------------------------------------------------------------------
__global__ __launch_bounds__(256) void k_topk(
    const float* __restrict__ score, const int* __restrict__ iterp,
    float* __restrict__ amin) {
  __shared__ __align__(16) float srow[NUMV];          // 8 KB
  __shared__ int red[256];
  int row = blockIdx.x;
  int t = threadIdx.x;

  if (t < 32) {  // wave 0: 1-D DMA of the whole score row into LDS
    const float* sp = score + (size_t)row * NUMV;
    tdm_load_2d((unsigned)(uintptr_t)(&srow[0]), sp,
                /*dsize=4B*/2, /*tdim0*/NUMV, /*tdim1*/1,
                /*tile0*/NUMV, /*tile1*/1, /*stride0*/NUMV);
    __builtin_amdgcn_s_wait_tensorcnt(0);
  }
  __syncthreads();

  int it = iterp[0];
  float kr = (float)NUMV * 0.1f * (float)(NUM_LAYER - 1 - it);
  int k = (int)floorf(kr + 0.5f);
  if (k < 1) k = 1;
  if (k > NUMV) k = NUMV;

  unsigned lo = 0u, hi = 0x7F800000u;   // count(>=lo)>=k, count(>=hi)<k
  for (int bi = 0; bi < 31; ++bi) {
    unsigned mid = lo + ((hi - lo) >> 1);
    float tv = __uint_as_float(mid);
    int c = 0;
    #pragma unroll
    for (int j = 0; j < NUMV / 256; ++j) c += (srow[t + j * 256] >= tv) ? 1 : 0;
    red[t] = c;
    __syncthreads();
    for (int s2 = 128; s2 > 0; s2 >>= 1) {
      if (t < s2) red[t] += red[t + s2];
      __syncthreads();
    }
    int total = red[0];
    __syncthreads();
    if (total >= k) lo = mid; else hi = mid;
  }
  if (t == 0) amin[row] = __uint_as_float(lo);
}

// ---------------------------------------------------------------------------
// 6) Row pass: W, H outputs + Dv = 1/(row-sum(H)+1e-10)
// ---------------------------------------------------------------------------
__global__ __launch_bounds__(256) void k_rows(
    const float* __restrict__ score, const float* __restrict__ amin,
    const float* __restrict__ H0, float* __restrict__ outH,
    float* __restrict__ outW, float* __restrict__ outDv) {
  __shared__ float red[256];
  int row = blockIdx.x;
  int t = threadIdx.x;
  float a = amin[row];
  size_t base = (size_t)row * NUMV;
  float rsum = 0.f;
  #pragma unroll
  for (int j = 0; j < NUMV / 256; ++j) {
    size_t idx = base + t + j * 256;
    float sc = score[idx];
    float w = (sc >= a) ? sc : 0.0f;
    float h = (w > 0.0f) ? H0[idx] : 0.0f;
    outW[idx] = w;
    outH[idx] = h;
    rsum += h;
  }
  red[t] = rsum; __syncthreads();
  for (int s2 = 128; s2 > 0; s2 >>= 1) {
    if (t < s2) red[t] += red[t + s2];
    __syncthreads();
  }
  if (t == 0) outDv[row] = 1.0f / (red[0] + 1e-10f);
}

// ---------------------------------------------------------------------------
// 7) Column pass (coalesced along i): De = 1/(col-sum(H)+1e-10), col-sum(W)
// ---------------------------------------------------------------------------
__global__ __launch_bounds__(256) void k_cols(
    const float* __restrict__ score, const float* __restrict__ amin,
    const float* __restrict__ H0, float* __restrict__ outDe,
    float* __restrict__ colw) {
  int b = blockIdx.y;
  int i = blockIdx.x * 256 + threadIdx.x;
  const float* sp = score + (size_t)b * NUMV * NUMV;
  const float* hp = H0    + (size_t)b * NUMV * NUMV;
  const float* ap = amin  + (size_t)b * NUMV;
  float ch = 0.f, cw = 0.f;
  for (int o = 0; o < NUMV; ++o) {
    float a = ap[o];
    float sc = sp[(size_t)o * NUMV + i];
    float w = (sc >= a) ? sc : 0.0f;
    float h = (w > 0.0f) ? hp[(size_t)o * NUMV + i] : 0.0f;
    ch += h; cw += w;
  }
  outDe[b * NUMV + i] = 1.0f / (ch + 1e-10f);
  colw[b * NUMV + i]  = cw;
}

// ---------------------------------------------------------------------------
// 8) W_edge = colsum(W) / max(||colsum(W)||_2, 1e-12) per batch
// ---------------------------------------------------------------------------
__global__ __launch_bounds__(256) void k_norm(
    const float* __restrict__ colw, float* __restrict__ outWe) {
  __shared__ float red[256];
  int b = blockIdx.x, t = threadIdx.x;
  const float* cp = colw + (size_t)b * NUMV;
  float ss = 0.f;
  #pragma unroll
  for (int j = 0; j < NUMV / 256; ++j) { float v = cp[t + j * 256]; ss += v * v; }
  red[t] = ss; __syncthreads();
  for (int s2 = 128; s2 > 0; s2 >>= 1) {
    if (t < s2) red[t] += red[t + s2];
    __syncthreads();
  }
  float nrm = fmaxf(sqrtf(red[0]), 1e-12f);
  #pragma unroll
  for (int j = 0; j < NUMV / 256; ++j)
    outWe[b * NUMV + t + j * 256] = cp[t + j * 256] / nrm;
}

// ---------------------------------------------------------------------------
extern "C" void kernel_launch(void* const* d_in, const int* in_sizes, int n_in,
                              void* d_out, int out_size, void* d_ws, size_t ws_size,
                              hipStream_t stream) {
  const float* H0    = (const float*)d_in[0];
  const float* vfeat = (const float*)d_in[1];
  const float* efeat = (const float*)d_in[2];
  const float* Wq    = (const float*)d_in[3];
  const float* bq    = (const float*)d_in[4];
  const float* Wk    = (const float*)d_in[5];
  const float* bk    = (const float*)d_in[6];
  const int*   iter  = (const int*)d_in[7];
  float* out = (float*)d_out;

  // Workspace layout (all 256B-aligned)
  char* ws = (char*)d_ws;
  size_t off = 0;
  auto take = [&](size_t bytes) -> char* {
    char* p = ws + off;
    off = (off + bytes + 255) & ~(size_t)255;
    return p;
  };
  const size_t featBytes = (size_t)BS * NUMV * CDIM * sizeof(bf16_t);
  bf16_t* vT   = (bf16_t*)take(featBytes);
  bf16_t* eT   = (bf16_t*)take(featBytes);
  bf16_t* QT   = (bf16_t*)take(featBytes);
  bf16_t* KT   = (bf16_t*)take(featBytes);
  bf16_t* Wqb  = (bf16_t*)take((size_t)CDIM * CDIM * sizeof(bf16_t));
  bf16_t* Wkb  = (bf16_t*)take((size_t)CDIM * CDIM * sizeof(bf16_t));
  float*  score= (float*) take((size_t)BS * NUMV * NUMV * sizeof(float));
  float*  amin = (float*) take((size_t)BS * NUMV * sizeof(float));
  float*  colw = (float*) take((size_t)BS * NUMV * sizeof(float));

  // Output slices (H, W, De, Dv, W_edge)
  const size_t BNN = (size_t)BS * NUMV * NUMV;
  float* outH  = out;
  float* outW  = out + BNN;
  float* outDe = out + 2 * BNN;
  float* outDv = out + 2 * BNN + (size_t)BS * NUMV;
  float* outWe = out + 2 * BNN + 2 * (size_t)BS * NUMV;

  k_transpose_bf16<<<dim3(NUMV / 32, CDIM / 32, BS * 2), dim3(32, 8, 1), 0, stream>>>(
      vfeat, efeat, vT, eT);
  k_convw<<<dim3((2 * CDIM * CDIM) / 256, 1, 1), dim3(256, 1, 1), 0, stream>>>(
      Wq, Wk, Wqb, Wkb);
  k_proj<<<dim3(NUMV / 16, CDIM / 128, BS * 2), dim3(256, 1, 1), 0, stream>>>(
      vT, eT, Wqb, Wkb, bq, bk, QT, KT);
  k_attn<<<dim3(NUMV / 16, NUMV / 128, BS), dim3(256, 1, 1), 0, stream>>>(
      QT, KT, H0, score);
  k_topk<<<dim3(BS * NUMV, 1, 1), dim3(256, 1, 1), 0, stream>>>(score, iter, amin);
  k_rows<<<dim3(BS * NUMV, 1, 1), dim3(256, 1, 1), 0, stream>>>(
      score, amin, H0, outH, outW, outDv);
  k_cols<<<dim3(NUMV / 256, BS, 1), dim3(256, 1, 1), 0, stream>>>(
      score, amin, H0, outDe, colw);
  k_norm<<<dim3(BS, 1, 1), dim3(256, 1, 1), 0, stream>>>(colw, outWe);
}
